// GNN_51213190037703
// MI455X (gfx1250) — matrix operand order repaired and checked by reading, not compiled, not run on hardware
//
#include <hip/hip_runtime.h>
#include <hip/hip_bf16.h>

// ---------------------------------------------------------------------------
// GCN 3-layer forward for MI455X (gfx1250, wave32, WMMA).
// GEMMs run on v_wmma_f32_16x16x32_bf16; message passing is float4 gather +
// global f32 atomic scatter (memory-bound part, ~2.6GB -> ~110us @ 23.3TB/s).
// ---------------------------------------------------------------------------

typedef __bf16 bf16_t;
typedef __attribute__((ext_vector_type(16))) __bf16 v16bf;
typedef __attribute__((ext_vector_type(8)))  __bf16 v8bf;
typedef __attribute__((ext_vector_type(8)))  float  v8f;

#define FIN   7
#define HID   128
#define OUTF  64
#define K1    32      // layer-1 K padded from 7 to 32 (one WMMA k-step)
#define NGRAPH 50

static __device__ __forceinline__ int gtid() {
    return blockIdx.x * blockDim.x + threadIdx.x;
}

// ------------------------------ small prep kernels -------------------------

__global__ void k_fill(float* p, float v, int n) {
    int i = gtid();
    if (i < n) p[i] = v;
}

__global__ void k_count_deg(const int* __restrict__ dst, float* __restrict__ deg, int e) {
    int i = gtid();
    if (i < e) atomicAdd(&deg[dst[i]], 1.0f);
}

__global__ void k_rsqrt_inplace(float* __restrict__ deg, int n) {
    int i = gtid();
    if (i < n) deg[i] = rsqrtf(deg[i]);
}

// x [N,7] f32 -> Xb [N,32] bf16 (zero padded)
__global__ void k_x_to_bf16(const float* __restrict__ x, bf16_t* __restrict__ Xb, int n) {
    int i = gtid();
    if (i >= n * K1) return;
    int node = i / K1, k = i % K1;
    float v = (k < FIN) ? x[node * FIN + k] : 0.0f;
    Xb[i] = (bf16_t)v;
}

// W1 [7,128] f32 -> W1t [128,32] bf16 (K padded)
__global__ void k_w1_transpose(const float* __restrict__ W1, bf16_t* __restrict__ W1t) {
    int i = gtid();
    if (i >= HID * K1) return;
    int nn = i / K1, k = i % K1;
    W1t[i] = (bf16_t)((k < FIN) ? W1[k * HID + nn] : 0.0f);
}

// W [K,F] f32 -> Wt [F,K] bf16
__global__ void k_w_transpose(const float* __restrict__ W, bf16_t* __restrict__ Wt, int K, int F) {
    int i = gtid();
    if (i >= K * F) return;
    int nn = i / K, k = i % K;
    Wt[i] = (bf16_t)W[k * F + nn];
}

// ------------------------------ WMMA GEMM ----------------------------------
// T[rows, FOUT] = A[rows, K](bf16) @ Bt[FOUT, K](bf16)^T, f32 accumulate.
// One wave per 16x16 output tile; blockDim.x = 32*(FOUT/16); grid.x = rows/16.
template <int FOUT, int K>
__global__ void k_gemm_wmma(const bf16_t* __restrict__ A,
                            const bf16_t* __restrict__ Bt,
                            float* __restrict__ T) {
    const int lane = threadIdx.x & 31;
    const int wave = threadIdx.x >> 5;
    const int m0   = blockIdx.x * 16;
    const int n0   = wave * 16;
    const int half = (lane >> 4) & 1;           // 0: lanes 0-15, 1: lanes 16-31
    const int mrow = m0 + (lane & 15);
    const int ncol = n0 + (lane & 15);

    const bf16_t* arow = A  + (size_t)mrow * K;
    const bf16_t* brow = Bt + (size_t)ncol * K;

    v8f acc = {};
#pragma unroll
    for (int kk = 0; kk < K; kk += 32) {
        // A 16x32 bf16 layout: lanes 0-15 hold K {0..7,16..23}; lanes 16-31 {8..15,24..31}
        v8bf alo = *(const v8bf*)(arow + kk + half * 8);
        v8bf ahi = *(const v8bf*)(arow + kk + half * 8 + 16);
        // B 32x16 bf16 layout: lane holds column n, K {0..7,16..23} (lanes 0-15) / {8..15,24..31}
        v8bf blo = *(const v8bf*)(brow + kk + half * 8);
        v8bf bhi = *(const v8bf*)(brow + kk + half * 8 + 16);
        v16bf a, b;
#pragma unroll
        for (int i = 0; i < 8; ++i) {
            a[i] = alo[i]; a[i + 8] = ahi[i];
            b[i] = blo[i]; b[i + 8] = bhi[i];
        }
        acc = __builtin_amdgcn_wmma_f32_16x16x32_bf16(
            false, a, false, b, (short)0, acc, false, false);
    }
    // C/D layout: VGPR j -> row m0 + j + half*8, col = n0 + (lane&15)
#pragma unroll
    for (int j = 0; j < 8; ++j) {
        int r = m0 + j + half * 8;
        T[(size_t)r * FOUT + ncol] = acc[j];
    }
}

// ------------------------------ edge scatter -------------------------------
// acc[dst] += T[src] * dinv[src]*dinv[dst], float4 per thread.
template <int F>
__global__ void k_scatter(const int* __restrict__ src, const int* __restrict__ dst,
                          const float* __restrict__ dinv, const float* __restrict__ T,
                          float* __restrict__ acc, int e) {
    const int TPE = F / 4;
    int gid = gtid();
    int ei  = gid / TPE;
    int c   = (gid % TPE) * 4;
    if (ei >= e) return;
    int s = src[ei], d = dst[ei];
    float w = dinv[s] * dinv[d];
    const float4 v = *(const float4*)(T + (size_t)s * F + c);
    float* o = acc + (size_t)d * F + c;
    atomicAdd(o + 0, v.x * w);
    atomicAdd(o + 1, v.y * w);
    atomicAdd(o + 2, v.z * w);
    atomicAdd(o + 3, v.w * w);
}

// ------------------------------ finalize -----------------------------------
// out = acc + T*dinv^2 (self-loop) + bias, optional ReLU; re-pack bf16 and/or f32.
template <int F, bool RELU>
__global__ void k_finalize(const float* __restrict__ acc, const float* __restrict__ T,
                           const float* __restrict__ dinv, const float* __restrict__ bias,
                           bf16_t* __restrict__ Hb, float* __restrict__ Hf, int n) {
    int i = gtid();
    if (i >= n * F) return;
    int node = i / F, f = i % F;
    float di = dinv[node];
    float v = acc[i] + T[i] * (di * di) + bias[f];
    if (RELU) v = fmaxf(v, 0.0f);
    if (Hb) Hb[i] = (bf16_t)v;
    if (Hf) Hf[i] = v;
}

// ------------------------------ pooling ------------------------------------

__global__ void k_pool_accum(const float* __restrict__ h, const int* __restrict__ batch,
                             float* __restrict__ sums, float* __restrict__ cnts, int n) {
    int i = gtid();
    if (i >= n * OUTF) return;
    int node = i / OUTF, f = i % OUTF;
    int g = batch[node];
    atomicAdd(&sums[g * OUTF + f], h[i]);
    if (f == 0) atomicAdd(&cnts[g], 1.0f);
}

__global__ void k_pool_div(const float* __restrict__ sums, const float* __restrict__ cnts,
                           float* __restrict__ out) {
    int i = gtid();
    if (i < NGRAPH * OUTF) out[i] = sums[i] / fmaxf(cnts[i / OUTF], 1.0f);
}

// ------------------------------ launcher -----------------------------------

static inline int cdiv(long long a, int b) { return (int)((a + b - 1) / b); }

extern "C" void kernel_launch(void* const* d_in, const int* in_sizes, int n_in,
                              void* d_out, int out_size, void* d_ws, size_t ws_size,
                              hipStream_t stream) {
    const float* x    = (const float*)d_in[0];
    const int*   eidx = (const int*)d_in[1];
    const int*   batch= (const int*)d_in[2];
    const float* W1   = (const float*)d_in[3];
    const float* b1   = (const float*)d_in[4];
    const float* W2   = (const float*)d_in[5];
    const float* b2   = (const float*)d_in[6];
    const float* W3   = (const float*)d_in[7];
    const float* b3   = (const float*)d_in[8];

    const int N = in_sizes[0] / FIN;     // 50000
    const int E = in_sizes[1] / 2;       // 800000
    const int* src = eidx;
    const int* dst = eidx + E;

    // ---- workspace carve (256B aligned) ----
    char* w = (char*)d_ws;
    auto carve = [&](size_t bytes) -> void* {
        void* p = (void*)w;
        w += (bytes + 255) & ~(size_t)255;
        return p;
    };
    float*  dinv = (float*) carve((size_t)N * 4);           // deg -> dinv
    float*  T    = (float*) carve((size_t)N * HID * 4);     // GEMM output
    float*  acc  = (float*) carve((size_t)N * HID * 4);     // scatter accumulator
    bf16_t* Hb   = (bf16_t*)carve((size_t)N * HID * 2);     // bf16 activations (also Xb)
    bf16_t* W1t  = (bf16_t*)carve((size_t)HID * K1  * 2);
    bf16_t* W2t  = (bf16_t*)carve((size_t)HID * HID * 2);
    bf16_t* W3t  = (bf16_t*)carve((size_t)OUTF * HID * 2);
    float*  sums = (float*) carve((size_t)NGRAPH * OUTF * 4);
    float*  cnts = (float*) carve((size_t)NGRAPH * 4);

    float* h_out = (float*)d_out;                 // [N, OUTF]
    float* p_out = h_out + (size_t)N * OUTF;      // [NGRAPH, OUTF]

    const int TB = 256;

    // ---- degree normalization ----
    k_fill<<<cdiv(N, TB), TB, 0, stream>>>(dinv, 1.0f, N);          // self-loops
    k_count_deg<<<cdiv(E, TB), TB, 0, stream>>>(dst, dinv, E);
    k_rsqrt_inplace<<<cdiv(N, TB), TB, 0, stream>>>(dinv, N);

    // ---- pack inputs / weights to bf16 ----
    k_x_to_bf16<<<cdiv((long long)N * K1, TB), TB, 0, stream>>>(x, Hb, N);
    k_w1_transpose<<<cdiv(HID * K1, TB), TB, 0, stream>>>(W1, W1t);
    k_w_transpose<<<cdiv(HID * HID, TB), TB, 0, stream>>>(W2, W2t, HID, HID);
    k_w_transpose<<<cdiv(HID * OUTF, TB), TB, 0, stream>>>(W3, W3t, HID, OUTF);

    const int mtiles = N / 16;  // 3125

    // ---- layer 1: [N,32]bf16 @ [32,128] -> relu -> Hb ----
    k_fill<<<cdiv((long long)N * HID, TB), TB, 0, stream>>>(acc, 0.0f, N * HID);
    k_gemm_wmma<HID, K1><<<mtiles, 32 * (HID / 16), 0, stream>>>(Hb, W1t, T);
    k_scatter<HID><<<cdiv((long long)E * (HID / 4), TB), TB, 0, stream>>>(src, dst, dinv, T, acc, E);
    k_finalize<HID, true><<<cdiv((long long)N * HID, TB), TB, 0, stream>>>(acc, T, dinv, b1, Hb, nullptr, N);

    // ---- layer 2: [N,128] @ [128,128] -> relu -> Hb ----
    k_fill<<<cdiv((long long)N * HID, TB), TB, 0, stream>>>(acc, 0.0f, N * HID);
    k_gemm_wmma<HID, HID><<<mtiles, 32 * (HID / 16), 0, stream>>>(Hb, W2t, T);
    k_scatter<HID><<<cdiv((long long)E * (HID / 4), TB), TB, 0, stream>>>(src, dst, dinv, T, acc, E);
    k_finalize<HID, true><<<cdiv((long long)N * HID, TB), TB, 0, stream>>>(acc, T, dinv, b2, Hb, nullptr, N);

    // ---- layer 3: [N,128] @ [128,64] -> h_out (f32, no relu) ----
    k_fill<<<cdiv((long long)N * OUTF, TB), TB, 0, stream>>>(acc, 0.0f, N * OUTF);
    k_gemm_wmma<OUTF, HID><<<mtiles, 32 * (OUTF / 16), 0, stream>>>(Hb, W3t, T);
    k_scatter<OUTF><<<cdiv((long long)E * (OUTF / 4), TB), TB, 0, stream>>>(src, dst, dinv, T, acc, E);
    k_finalize<OUTF, false><<<cdiv((long long)N * OUTF, TB), TB, 0, stream>>>(acc, T, dinv, b3, nullptr, h_out, N);

    // ---- global mean pool ----
    k_fill<<<cdiv(NGRAPH * OUTF, TB), TB, 0, stream>>>(sums, 0.0f, NGRAPH * OUTF);
    k_fill<<<1, TB, 0, stream>>>(cnts, 0.0f, NGRAPH);
    k_pool_accum<<<cdiv((long long)N * OUTF, TB), TB, 0, stream>>>(h_out, batch, sums, cnts, N);
    k_pool_div<<<cdiv(NGRAPH * OUTF, TB), TB, 0, stream>>>(sums, cnts, p_out);
}
